// LayerNormMLP_81570018885755
// MI455X (gfx1250) — matrix operand order repaired and checked by reading, not compiled, not run on hardware
//
#include <hip/hip_runtime.h>
#include <stdint.h>

// ---------------------------------------------------------------------------
// Problem constants (match reference exactly)
// ---------------------------------------------------------------------------
static constexpr int Bn = 4;
static constexpr int S  = 2048;
static constexpr int M  = Bn * S;   // 8192 tokens
static constexpr int H  = 2048;     // hidden
static constexpr int I  = 8192;     // intermediate
static constexpr float LN_EPS = 1e-6f;

typedef __attribute__((ext_vector_type(16))) __bf16 v16bf;
typedef __attribute__((ext_vector_type(8)))  float  v8f;
typedef __attribute__((ext_vector_type(4)))  int    v4i;

union FragAB { uint4 u[2]; v16bf v; };

// ---------------------------------------------------------------------------
// CDNA5 async global->LDS copy (ASYNCcnt-tracked). Fall back to sync copy if
// the builtin is unavailable on this toolchain.
// ---------------------------------------------------------------------------
#if defined(__has_builtin)
# if __has_builtin(__builtin_amdgcn_global_load_async_to_lds_b128)
#  define ASYNC_LDS 1
# endif
#endif
#ifndef ASYNC_LDS
# define ASYNC_LDS 0
#endif

typedef __attribute__((address_space(1))) v4i gv4i;   // global int4
typedef __attribute__((address_space(3))) v4i lv4i;   // LDS int4

__device__ __forceinline__ void async_cp16(const __bf16* g, __bf16* l) {
#if ASYNC_LDS
    // generic LDS pointer carries the LDS byte offset in its low 32 bits
    __builtin_amdgcn_global_load_async_to_lds_b128(
        (gv4i*)(uintptr_t)g,
        (lv4i*)(unsigned int)(uintptr_t)l,
        0, 0);
#else
    *(uint4*)l = *(const uint4*)g;
#endif
}

__device__ __forceinline__ void async_wait_all() {
#if ASYNC_LDS
    asm volatile("s_wait_asynccnt 0x0" ::: "memory");
#endif
}

// threshold wait: async loads complete in order, so waiting ASYNCcnt <= N
// with N = loads-per-stage guarantees the oldest outstanding stage is done
// while the newest stays in flight.
template <int N>
__device__ __forceinline__ void async_wait_le() {
#if ASYNC_LDS
    asm volatile("s_wait_asynccnt %0" :: "n"(N) : "memory");
#endif
}

__device__ __forceinline__ __bf16 f2bf(float f) {
    union { float f; unsigned u; } x; x.f = f;
    unsigned r = x.u + 0x7FFFu + ((x.u >> 16) & 1u);   // round-to-nearest-even
    union { unsigned short s; __bf16 b; } y;
    y.s = (unsigned short)(r >> 16);
    return y.b;
}

__device__ __forceinline__ float gelu_tanh(float y) {
    float c = 0.7978845608028654f * (y + 0.044715f * y * y * y);
    return 0.5f * y * (1.0f + tanhf(c));
}

// ---------------------------------------------------------------------------
// Kernel 1: LayerNorm over H, fp32 -> bf16.  One block (256 threads) per row.
// ---------------------------------------------------------------------------
__global__ __launch_bounds__(256)
void ln_bf16_kernel(const float* __restrict__ x,
                    const float* __restrict__ scale,
                    const float* __restrict__ bias,
                    __bf16* __restrict__ out) {
    const int row = blockIdx.x;
    const float* xr = x + (size_t)row * H;
    float v[8];
    float s = 0.f, sq = 0.f;
#pragma unroll
    for (int i = 0; i < 8; ++i) {
        float t = xr[threadIdx.x + i * 256];
        v[i] = t; s += t; sq += t * t;
    }
    __shared__ float rs[256], rq[256];
    rs[threadIdx.x] = s; rq[threadIdx.x] = sq;
    __syncthreads();
    for (int off = 128; off > 0; off >>= 1) {
        if ((int)threadIdx.x < off) {
            rs[threadIdx.x] += rs[threadIdx.x + off];
            rq[threadIdx.x] += rq[threadIdx.x + off];
        }
        __syncthreads();
    }
    const float mean = rs[0] * (1.0f / H);
    const float var  = rq[0] * (1.0f / H) - mean * mean;
    const float inv  = rsqrtf(var + LN_EPS);
    __bf16* orow = out + (size_t)row * H;
#pragma unroll
    for (int i = 0; i < 8; ++i) {
        int hh = threadIdx.x + i * 256;
        orow[hh] = f2bf((v[i] - mean) * inv * scale[hh] + bias[hh]);
    }
}

// ---------------------------------------------------------------------------
// Kernel 2: tiled transpose + fp32 -> bf16.  in: [R][C] fp32, out: [C][R] bf16
// ---------------------------------------------------------------------------
__global__ __launch_bounds__(256)
void transpose_cvt_kernel(const float* __restrict__ in,
                          __bf16* __restrict__ out,
                          int R, int C) {
    __shared__ float tile[32][33];
    const int c0 = blockIdx.x * 32;
    const int r0 = blockIdx.y * 32;
    const int tx = threadIdx.x, ty = threadIdx.y;
#pragma unroll
    for (int k = 0; k < 4; ++k)
        tile[ty + k * 8][tx] = in[(size_t)(r0 + ty + k * 8) * C + (c0 + tx)];
    __syncthreads();
#pragma unroll
    for (int k = 0; k < 4; ++k)
        out[(size_t)(c0 + ty + k * 8) * R + (r0 + tx)] = f2bf(tile[tx][ty + k * 8]);
}

// ---------------------------------------------------------------------------
// WMMA fragment loaders (bf16 16x16x32, wave32 layouts per CDNA5 ISA 7.12.2)
// A: row M = lane%16; per-lane K runs [h*8, h*8+8) and [16+h*8, 16+h*8+8)
// B: stored transposed in LDS as [n][k]; col N = lane%16; K runs [h*8, h*8+8) and [16+h*8, 16+h*8+8)
// ---------------------------------------------------------------------------
static constexpr int LSTR = 72;   // LDS row stride in bf16 (144 B) - bank spread

__device__ __forceinline__ v16bf load_a_frag(const __bf16* lds, int mbase, int kk, int lane) {
    const int m = mbase + (lane & 15);
    const int h = lane >> 4;
    const __bf16* p = lds + m * LSTR + kk + h * 8;
    FragAB f;
    f.u[0] = *(const uint4*)(p);
    f.u[1] = *(const uint4*)(p + 16);
    return f.v;
}

__device__ __forceinline__ v16bf load_b_frag(const __bf16* lds, int nbase, int kk, int lane) {
    const int n = nbase + (lane & 15);
    const int h = lane >> 4;
    const __bf16* p = lds + n * LSTR + kk + h * 8;
    FragAB f;
    f.u[0] = *(const uint4*)(p);
    f.u[1] = *(const uint4*)(p + 16);
    return f.v;
}

// ---------------------------------------------------------------------------
// Kernel 3/4: async triple-buffered WMMA GEMM.
//   A  : [M][K] bf16 row-major;  Bt : [Ntot][K] bf16 (row n = output column n)
//   GATED: acc1 from Bt rows [NOUT+n0..); epilogue z = gelu(y0)*y1 -> bf16
// Block tile 128(M) x 64(N), 8 waves 4x2, wave tile 32x32, K staged 64/iter.
// 3-stage async pipeline with the drain peeled out of the steady-state loop.
// ---------------------------------------------------------------------------
template <bool GATED>
__global__ __launch_bounds__(256)
void gemm_wmma_kernel(const __bf16* __restrict__ A,
                      const __bf16* __restrict__ Bt,
                      void* __restrict__ Out,
                      int K, int NOUT) {
    __shared__ __bf16 lA [3][128 * LSTR];
    __shared__ __bf16 lB0[3][ 64 * LSTR];
    __shared__ __bf16 lB1[GATED ? 3 : 1][64 * LSTR];

    constexpr int NPER = GATED ? 8 : 6;   // async loads issued per stage per wave

    const int tid  = threadIdx.x;
    const int lane = tid & 31;
    const int wid  = tid >> 5;
    const int wm   = (wid & 3) * 32;
    const int wn   = (wid >> 2) * 32;

    const int n0 = blockIdx.x * 64;
    const int m0 = blockIdx.y * 128;

    // per-thread staging job: row r (0..31), 16B chunk c (0..7); A covers rows
    // r,r+32,r+64,r+96 of the 128-row tile, B covers rows r,r+32 of 64.
    const int r = tid >> 3, c = tid & 7;
    const size_t rstep = (size_t)32 * K;                  // 32 rows in elements
    const __bf16* gA  = A  + (size_t)(m0 + r) * K + c * 8;
    const __bf16* gB0 = Bt + (size_t)(n0 + r) * K + c * 8;
    const __bf16* gB1 = Bt + (size_t)((GATED ? NOUT : 0) + n0 + r) * K + c * 8;
    const int lofs = r * LSTR + c * 8;

    auto issue_stage = [&](int buf, int ks) {
        const __bf16* a = gA + ks;
        __bf16* la = lA[buf];
#pragma unroll
        for (int i = 0; i < 4; ++i)
            async_cp16(a + (size_t)i * rstep, la + lofs + i * 32 * LSTR);
        const __bf16* b = gB0 + ks;
        __bf16* lb = lB0[buf];
#pragma unroll
        for (int i = 0; i < 2; ++i)
            async_cp16(b + (size_t)i * rstep, lb + lofs + i * 32 * LSTR);
        if (GATED) {
            const __bf16* g2 = gB1 + ks;
            __bf16* lg = lB1[buf];
#pragma unroll
            for (int i = 0; i < 2; ++i)
                async_cp16(g2 + (size_t)i * rstep, lg + lofs + i * 32 * LSTR);
        }
    };

    v8f acc0[2][2] = {};
    v8f acc1[2][2] = {};

    auto compute_stage = [&](int buf) {
        const __bf16* bA  = lA [buf];
        const __bf16* bB0 = lB0[buf];
        const __bf16* bB1 = lB1[GATED ? buf : 0];
#pragma unroll
        for (int kk = 0; kk < 64; kk += 32) {
            v16bf a0 = load_a_frag(bA, wm,      kk, lane);
            v16bf a1 = load_a_frag(bA, wm + 16, kk, lane);
            v16bf b0 = load_b_frag(bB0, wn,      kk, lane);
            v16bf b1 = load_b_frag(bB0, wn + 16, kk, lane);
            acc0[0][0] = __builtin_amdgcn_wmma_f32_16x16x32_bf16(false, a0, false, b0, (short)0, acc0[0][0], false, false);
            acc0[0][1] = __builtin_amdgcn_wmma_f32_16x16x32_bf16(false, a0, false, b1, (short)0, acc0[0][1], false, false);
            acc0[1][0] = __builtin_amdgcn_wmma_f32_16x16x32_bf16(false, a1, false, b0, (short)0, acc0[1][0], false, false);
            acc0[1][1] = __builtin_amdgcn_wmma_f32_16x16x32_bf16(false, a1, false, b1, (short)0, acc0[1][1], false, false);
            if (GATED) {
                v16bf c0 = load_b_frag(bB1, wn,      kk, lane);
                v16bf c1 = load_b_frag(bB1, wn + 16, kk, lane);
                acc1[0][0] = __builtin_amdgcn_wmma_f32_16x16x32_bf16(false, a0, false, c0, (short)0, acc1[0][0], false, false);
                acc1[0][1] = __builtin_amdgcn_wmma_f32_16x16x32_bf16(false, a0, false, c1, (short)0, acc1[0][1], false, false);
                acc1[1][0] = __builtin_amdgcn_wmma_f32_16x16x32_bf16(false, a1, false, c0, (short)0, acc1[1][0], false, false);
                acc1[1][1] = __builtin_amdgcn_wmma_f32_16x16x32_bf16(false, a1, false, c1, (short)0, acc1[1][1], false, false);
            }
        }
    };

    const int nstages = K >> 6;           // K / 64  (>= 32 for our shapes)
    issue_stage(0, 0);
    issue_stage(1, 64);

    // steady state: stages [0, nstages-2) -- always prefetch two ahead
    int cbuf = 0;
    for (int s = 0; s < nstages - 2; ++s) {
        async_wait_le<NPER>();            // stage s resident; s+1 in flight
        __syncthreads();
        int ibuf = cbuf - 1; if (ibuf < 0) ibuf += 3;   // (s+2) % 3
        issue_stage(ibuf, (s + 2) << 6);
        compute_stage(cbuf);
        cbuf = (cbuf == 2) ? 0 : cbuf + 1;
    }
    // stage nstages-2: last stage still in flight
    async_wait_le<NPER>();
    __syncthreads();
    compute_stage(cbuf);
    cbuf = (cbuf == 2) ? 0 : cbuf + 1;
    // stage nstages-1: drain
    async_wait_all();
    __syncthreads();
    compute_stage(cbuf);

    // epilogue: C/D layout -> element (m = v + 8*h, n = lane%16) per 16x16 tile
    const int h  = lane >> 4;
    const int nn = lane & 15;
#pragma unroll
    for (int i = 0; i < 2; ++i)
#pragma unroll
        for (int j = 0; j < 2; ++j)
#pragma unroll
            for (int v = 0; v < 8; ++v) {
                const int gm = m0 + wm + i * 16 + v + 8 * h;
                const int gn = n0 + wn + j * 16 + nn;
                if (GATED) {
                    float y0 = acc0[i][j][v];
                    float y1 = acc1[i][j][v];
                    ((__bf16*)Out)[(size_t)gm * NOUT + gn] = f2bf(gelu_tanh(y0) * y1);
                } else {
                    ((float*)Out)[(size_t)gm * NOUT + gn] = acc0[i][j][v];
                }
            }
}

// ---------------------------------------------------------------------------
// launch
// ---------------------------------------------------------------------------
extern "C" void kernel_launch(void* const* d_in, const int* in_sizes, int n_in,
                              void* d_out, int out_size, void* d_ws, size_t ws_size,
                              hipStream_t stream) {
    const float* x      = (const float*)d_in[0];   // [B,S,H]
    const float* scale  = (const float*)d_in[1];   // [H]
    const float* lnbias = (const float*)d_in[2];   // [H]
    const float* k1     = (const float*)d_in[3];   // [H][2I]
    const float* k2     = (const float*)d_in[4];   // [I][H]
    float* out          = (float*)d_out;           // [B,S,H] fp32

    char* ws = (char*)d_ws;
    __bf16* lnb = (__bf16*)(ws);                               // [M][H]     32 MiB
    __bf16* k1t = (__bf16*)(ws + (size_t)33554432);            // [2I][H]    64 MiB
    __bf16* k2t = (__bf16*)(ws + (size_t)100663296);           // [H][I]     32 MiB
    __bf16* z   = (__bf16*)(ws + (size_t)134217728);           // [M][I]    128 MiB

    ln_bf16_kernel<<<M, 256, 0, stream>>>(x, scale, lnbias, lnb);

    transpose_cvt_kernel<<<dim3((2 * I) / 32, H / 32), dim3(32, 8), 0, stream>>>(k1, k1t, H, 2 * I);
    transpose_cvt_kernel<<<dim3(H / 32, I / 32),       dim3(32, 8), 0, stream>>>(k2, k2t, I, H);

    gemm_wmma_kernel<true><<<dim3(I / 64, M / 128), 256, 0, stream>>>(lnb, k1t, (void*)z, H, I);
    gemm_wmma_kernel<false><<<dim3(H / 64, M / 128), 256, 0, stream>>>(z, k2t, (void*)out, I, H);
}
